// LRUEmbedding_43310450213190
// MI455X (gfx1250) — matrix operand order repaired and checked by reading, not compiled, not run on hardware
//
#include <hip/hip_runtime.h>
#include <hip/hip_bf16.h>
#include <math.h>

typedef __bf16 bh16;
typedef __attribute__((ext_vector_type(16))) __bf16 v16bf;
typedef __attribute__((ext_vector_type(8)))  float  v8f;

#define BATCHN 16
#define LSEQ   2048
#define MTOT   (BATCHN*LSEQ)   // 32768 rows
#define INDIM  64
#define HID    256
#define STATE  256
#define MLPD   1024
#define OUTD   128
#define NL     4
#define CHUNKS 64
#define CLEN   (LSEQ/CHUNKS)   // 32

union ABu { v16bf v; uint4 q[2]; unsigned u[8]; };

// ---- CDNA5 async global->LDS copy (ASYNCcnt path), with sync fallback ----
#if __has_builtin(__builtin_amdgcn_global_load_async_to_lds_b128)
#define HAVE_ASYNC 1
typedef int v4i_async __attribute__((vector_size(16)));
typedef __attribute__((address_space(1))) v4i_async gv4i;
typedef __attribute__((address_space(3))) v4i_async lv4i;
#if __has_builtin(__builtin_amdgcn_s_wait_asynccnt)
#define WAIT_ASYNC(n) __builtin_amdgcn_s_wait_asynccnt(n)
#else
#define WAIT_ASYNC(n) asm volatile("s_wait_asynccnt %0" ::"n"(n) : "memory")
#endif
#else
#define HAVE_ASYNC 0
#define WAIT_ASYNC(n)
#endif

__device__ __forceinline__ void cp_tile16(const bh16* __restrict__ g, bh16* l) {
#if HAVE_ASYNC
  __builtin_amdgcn_global_load_async_to_lds_b128((gv4i*)g, (lv4i*)l, 0, 0);
  __builtin_amdgcn_global_load_async_to_lds_b128((gv4i*)(g + 8), (lv4i*)(l + 8), 0, 0);
#else
  *(uint4*)l       = *(const uint4*)g;
  *(uint4*)(l + 8) = *(const uint4*)(g + 8);
#endif
}

// ---------------- small prep kernels ----------------

__global__ __launch_bounds__(256) void k_cvt_bf16(const float* __restrict__ s,
                                                  bh16* __restrict__ d, int n) {
  int i = blockIdx.x * blockDim.x + threadIdx.x;
  int stride = gridDim.x * blockDim.x;
  for (; i < n; i += stride) d[i] = (bh16)s[i];
}

__global__ __launch_bounds__(256) void k_prep_lam(const float* __restrict__ nu,
                                                  const float* __restrict__ th,
                                                  float2* __restrict__ lam) {
  int i = blockIdx.x * blockDim.x + threadIdx.x;
  if (i < NL * STATE) {
    float r = expf(-expf(nu[i]));
    float a = expf(th[i]);
    lam[i] = make_float2(r * cosf(a), r * sinf(a));
  }
}

// Bcat: per layer 512x256 (N x K) bf16: rows 0..255 = gamma*B_re, 256..511 = gamma*B_im
__global__ __launch_bounds__(256) void k_prep_Bcat(const float* __restrict__ Bre,
                                                   const float* __restrict__ Bim,
                                                   const float* __restrict__ gl,
                                                   bh16* __restrict__ out) {
  int i = blockIdx.x * blockDim.x + threadIdx.x;
  int n = NL * 2 * STATE * HID;
  if (i >= n) return;
  int k = i % HID;
  int r = (i / HID) % (2 * STATE);
  int l = i / (HID * 2 * STATE);
  int s = (r < STATE) ? r : r - STATE;
  float g = expf(gl[l * STATE + s]);
  const float* src = (r < STATE) ? Bre : Bim;
  out[i] = (bh16)(src[((size_t)l * STATE + s) * HID + k] * g);
}

// Ccat: per layer 256x512 (N x K) bf16: cols 0..255 = C_re, 256..511 = -C_im
__global__ __launch_bounds__(256) void k_prep_Ccat(const float* __restrict__ Cre,
                                                   const float* __restrict__ Cim,
                                                   bh16* __restrict__ out) {
  int i = blockIdx.x * blockDim.x + threadIdx.x;
  int n = NL * HID * 2 * STATE;
  if (i >= n) return;
  int k = i % (2 * STATE);
  int h = (i / (2 * STATE)) % HID;
  int l = i / (2 * STATE * HID);
  float v = (k < STATE) ? Cre[((size_t)l * HID + h) * STATE + k]
                        : -Cim[((size_t)l * HID + h) * STATE + (k - STATE)];
  out[i] = (bh16)v;
}

// ---------------- WMMA GEMM: out[M x N] = act( A[M x K]*W[N x K]^T + bias + dscale.*res )
// block = 256 threads = 8 waves; block tile 128x128; wave tile 32x64 (2x4 WMMA 16x16x32 bf16).
// Double-buffered LDS, async global->LDS copies when available.

__global__ __launch_bounds__(256) void k_gemm_wmma(
    const bh16* __restrict__ A, const bh16* __restrict__ W,
    const float* __restrict__ bias, const float* __restrict__ dscale,
    const bh16* __restrict__ res,
    float* __restrict__ outF, bh16* __restrict__ outB,
    int N, int K, int act)
{
  __shared__ __attribute__((aligned(16))) bh16 As[2][128 * 32];
  __shared__ __attribute__((aligned(16))) bh16 Bs[2][128 * 32];

  const int tid  = threadIdx.x;
  const int lane = tid & 31, wv = tid >> 5;
  const int wm = wv & 3, wn = wv >> 2;           // 4 waves in M, 2 in N
  const int bm = blockIdx.x * 128, bn = blockIdx.y * 128;
  const int m16 = lane & 15, half = lane >> 4;

  v8f acc[2][4] = {};

  // Each 128x32 tile: 256 threads x 16 bf16 (two b128 chunks).
  const int tr = tid >> 1, tc = (tid & 1) << 4;
  const bh16* gA0 = A + (size_t)(bm + tr) * K + tc;
  const bh16* gB0 = W + (size_t)(bn + tr) * K + tc;
  const int lofs = tr * 32 + tc;

  const int nk = K >> 5;
  // prologue: start buffer 0
  cp_tile16(gA0, &As[0][lofs]);
  cp_tile16(gB0, &Bs[0][lofs]);

  for (int kt = 0; kt < nk; ++kt) {
    const int cur = kt & 1;
    if (kt + 1 < nk) {
      cp_tile16(gA0 + ((kt + 1) << 5), &As[cur ^ 1][lofs]);
      cp_tile16(gB0 + ((kt + 1) << 5), &Bs[cur ^ 1][lofs]);
      WAIT_ASYNC(4);     // previous buffer's 4 copies complete; 4 new in flight
    } else {
      WAIT_ASYNC(0);
    }
    __syncthreads();

    const bh16* Ab = &As[cur][0];
    const bh16* Bb = &Bs[cur][0];
    ABu af[2], bf[4];
#pragma unroll
    for (int i = 0; i < 2; ++i) {
      // 16-bit A 16x32 layout: low lanes K0..7 & 16..23, high lanes K8..15 & 24..31
      int row = (wm * 32 + i * 16 + m16) * 32;
      af[i].q[0] = *(const uint4*)&Ab[row + 8 * half];
      af[i].q[1] = *(const uint4*)&Ab[row + 16 + 8 * half];
    }
#pragma unroll
    for (int j = 0; j < 4; ++j) {
      // 16-bit B 32x16 layout: low lanes K0..7 & 16..23, high lanes K8..15 & 24..31 (row n = lane&15)
      int row = (wn * 64 + j * 16 + m16) * 32;
      bf[j].q[0] = *(const uint4*)&Bb[row + 8 * half];
      bf[j].q[1] = *(const uint4*)&Bb[row + 16 + 8 * half];
    }
#pragma unroll
    for (int i = 0; i < 2; ++i)
#pragma unroll
      for (int j = 0; j < 4; ++j)
        acc[i][j] = __builtin_amdgcn_wmma_f32_16x16x32_bf16(
            false, af[i].v, false, bf[j].v, (short)0, acc[i][j], false, false);
    __syncthreads();   // all waves done reading buf[cur] before it is refilled
  }

  // Epilogue. C/D layout: lane n = lane&15; VGPR v holds row v (low half) / 8+v (high half).
#pragma unroll
  for (int i = 0; i < 2; ++i) {
    int gm0 = bm + wm * 32 + i * 16 + 8 * half;
#pragma unroll
    for (int j = 0; j < 4; ++j) {
      int gn = bn + wn * 64 + j * 16 + m16;
      float bv = bias ? bias[gn] : 0.0f;
      float ds = dscale ? dscale[gn] : 1.0f;
#pragma unroll
      for (int v = 0; v < 8; ++v) {
        size_t o = (size_t)(gm0 + v) * N + gn;
        float val = acc[i][j][v] + bv;
        if (res) val += ds * (float)res[o];
        if (act) val = 0.5f * val * (1.0f + erff(val * 0.70710678118654752f));
        if (outF) outF[o] = val;
        else      outB[o] = (bh16)val;
      }
    }
  }
}

// ---------------- chunked complex diagonal scan ----------------
// states layout matches Bu layout: (b, t, 2*STATE) with re at s, im at s+STATE.

__global__ __launch_bounds__(STATE) void k_scan_local(const bh16* __restrict__ Bu,
                                                      bh16* __restrict__ st,
                                                      float2* __restrict__ carry,
                                                      const float2* __restrict__ lam) {
  int b = blockIdx.x, c = blockIdx.y, s = threadIdx.x;
  float lr = lam[s].x, li = lam[s].y;
  float re = 0.f, im = 0.f;
  for (int tt = 0; tt < CLEN; ++tt) {
    size_t o = ((size_t)(b * LSEQ + c * CLEN + tt)) * (2 * STATE) + s;
    float br = (float)Bu[o], bi = (float)Bu[o + STATE];
    float nr = lr * re - li * im + br;
    float ni = lr * im + li * re + bi;
    re = nr; im = ni;
    st[o] = (bh16)re; st[o + STATE] = (bh16)im;
  }
  carry[((size_t)b * CHUNKS + c) * STATE + s] = make_float2(re, im);
}

__global__ __launch_bounds__(STATE) void k_scan_prefix(const float2* __restrict__ carry,
                                                       float2* __restrict__ pre,
                                                       const float2* __restrict__ lam) {
  int b = blockIdx.x, s = threadIdx.x;
  float ar = lam[s].x, ai = lam[s].y;
  for (int q = 0; q < 5; ++q) {      // alpha = lam^32
    float nr = ar * ar - ai * ai, ni = 2.f * ar * ai;
    ar = nr; ai = ni;
  }
  float pr = 0.f, pi = 0.f;
  for (int c = 0; c < CHUNKS; ++c) {
    size_t o = ((size_t)b * CHUNKS + c) * STATE + s;
    pre[o] = make_float2(pr, pi);
    float2 cv = carry[o];
    float nr = ar * pr - ai * pi + cv.x;
    float ni = ar * pi + ai * pr + cv.y;
    pr = nr; pi = ni;
  }
}

__global__ __launch_bounds__(STATE) void k_scan_fix(bh16* __restrict__ st,
                                                    const float2* __restrict__ pre,
                                                    const float2* __restrict__ lam) {
  int b = blockIdx.x, c = blockIdx.y, s = threadIdx.x;
  if (c == 0) return;                       // zero prefix
  float lr = lam[s].x, li = lam[s].y;
  float2 p = pre[((size_t)b * CHUNKS + c) * STATE + s];
  float pr = p.x, pi = p.y;
  for (int tt = 0; tt < CLEN; ++tt) {
    float nr = lr * pr - li * pi, ni = lr * pi + li * pr;
    pr = nr; pi = ni;
    size_t o = ((size_t)(b * LSEQ + c * CLEN + tt)) * (2 * STATE) + s;
    st[o]         = (bh16)((float)st[o] + pr);
    st[o + STATE] = (bh16)((float)st[o + STATE] + pi);
  }
}

// ---------------- driver ----------------

extern "C" void kernel_launch(void* const* d_in, const int* in_sizes, int n_in,
                              void* d_out, int out_size, void* d_ws, size_t ws_size,
                              hipStream_t stream)
{
  const float* x      = (const float*)d_in[0];
  const float* emb_W  = (const float*)d_in[1];
  const float* emb_b  = (const float*)d_in[2];
  const float* nu_log = (const float*)d_in[3];
  const float* th_log = (const float*)d_in[4];
  const float* ga_log = (const float*)d_in[5];
  const float* B_re   = (const float*)d_in[6];
  const float* B_im   = (const float*)d_in[7];
  const float* C_re   = (const float*)d_in[8];
  const float* C_im   = (const float*)d_in[9];
  const float* Dvec   = (const float*)d_in[10];
  const float* Wh     = (const float*)d_in[11];
  const float* bhv    = (const float*)d_in[12];
  const float* Wo     = (const float*)d_in[13];
  const float* bov    = (const float*)d_in[14];
  const float* out_W  = (const float*)d_in[15];
  const float* out_b  = (const float*)d_in[16];
  float* out = (float*)d_out;

  char* p = (char*)d_ws;
  auto alloc = [&](size_t bytes) {
    char* r = p;
    p += (bytes + 255) & ~(size_t)255;
    return r;
  };
  bh16* xB    = (bh16*)alloc((size_t)MTOT * INDIM * 2);
  bh16* hB    = (bh16*)alloc((size_t)MTOT * HID * 2);
  bh16* yB    = (bh16*)alloc((size_t)MTOT * HID * 2);
  bh16* BuB   = (bh16*)alloc((size_t)MTOT * 2 * STATE * 2);
  bh16* stB   = (bh16*)alloc((size_t)MTOT * 2 * STATE * 2);
  bh16* zB    = (bh16*)alloc((size_t)MTOT * MLPD * 2);
  bh16* embWB = (bh16*)alloc((size_t)HID * INDIM * 2);
  bh16* BcatB = (bh16*)alloc((size_t)NL * 2 * STATE * HID * 2);
  bh16* CcatB = (bh16*)alloc((size_t)NL * HID * 2 * STATE * 2);
  bh16* WhB   = (bh16*)alloc((size_t)NL * MLPD * HID * 2);
  bh16* WoB   = (bh16*)alloc((size_t)NL * HID * MLPD * 2);
  bh16* outWB = (bh16*)alloc((size_t)OUTD * HID * 2);
  float2* lamB = (float2*)alloc((size_t)NL * STATE * sizeof(float2));
  float2* carB = (float2*)alloc((size_t)BATCHN * CHUNKS * STATE * sizeof(float2));
  float2* preB = (float2*)alloc((size_t)BATCHN * CHUNKS * STATE * sizeof(float2));

  auto cvt = [&](const float* s, bh16* d, int n) {
    int blocks = (n + 255) / 256; if (blocks > 2048) blocks = 2048;
    hipLaunchKernelGGL(k_cvt_bf16, dim3(blocks), dim3(256), 0, stream, s, d, n);
  };
  cvt(x, xB, MTOT * INDIM);
  cvt(emb_W, embWB, HID * INDIM);
  cvt(Wh, WhB, NL * MLPD * HID);
  cvt(Wo, WoB, NL * HID * MLPD);
  cvt(out_W, outWB, OUTD * HID);
  hipLaunchKernelGGL(k_prep_lam, dim3((NL * STATE + 255) / 256), dim3(256), 0, stream,
                     nu_log, th_log, lamB);
  { int n = NL * 2 * STATE * HID;
    hipLaunchKernelGGL(k_prep_Bcat, dim3((n + 255) / 256), dim3(256), 0, stream,
                       B_re, B_im, ga_log, BcatB); }
  { int n = NL * HID * 2 * STATE;
    hipLaunchKernelGGL(k_prep_Ccat, dim3((n + 255) / 256), dim3(256), 0, stream,
                       C_re, C_im, CcatB); }

  auto gemm = [&](const bh16* A, const bh16* W_, const float* bias, const float* dsc,
                  const bh16* res, float* oF, bh16* oB, int N, int K, int act) {
    hipLaunchKernelGGL(k_gemm_wmma, dim3(MTOT / 128, N / 128), dim3(256), 0, stream,
                       A, W_, bias, dsc, res, oF, oB, N, K, act);
  };

  // h = x @ embW^T + emb_b
  gemm(xB, embWB, emb_b, nullptr, nullptr, nullptr, hB, HID, INDIM, 0);

  for (int l = 0; l < NL; ++l) {
    // Bu(re||im) = h @ Bcat^T
    gemm(hB, BcatB + (size_t)l * 2 * STATE * HID, nullptr, nullptr, nullptr, nullptr,
         BuB, 2 * STATE, HID, 0);
    // diagonal complex scan over L (chunk-parallel)
    hipLaunchKernelGGL(k_scan_local,  dim3(BATCHN, CHUNKS), dim3(STATE), 0, stream,
                       BuB, stB, carB, lamB + l * STATE);
    hipLaunchKernelGGL(k_scan_prefix, dim3(BATCHN),         dim3(STATE), 0, stream,
                       carB, preB, lamB + l * STATE);
    hipLaunchKernelGGL(k_scan_fix,    dim3(BATCHN, CHUNKS), dim3(STATE), 0, stream,
                       stB, preB, lamB + l * STATE);
    // y = Re(states @ C^T) + D*h   == [st_re||st_im] @ [C_re, -C_im]^T + D.*h
    gemm(stB, CcatB + (size_t)l * HID * 2 * STATE, nullptr, Dvec + l * HID, hB,
         nullptr, yB, HID, 2 * STATE, 0);
    // z = gelu(y @ Wh^T + bh)
    gemm(yB, WhB + (size_t)l * MLPD * HID, bhv + l * MLPD, nullptr, nullptr, nullptr,
         zB, MLPD, HID, 1);
    // h = z @ Wo^T + bo + y
    gemm(zB, WoB + (size_t)l * HID * MLPD, bov + l * HID, nullptr, yB, nullptr,
         hB, HID, MLPD, 0);
  }

  // out = h @ outW^T + out_b  (f32 output)
  gemm(hB, outWB, out_b, nullptr, nullptr, out, nullptr, OUTD, HID, 0);
}
